// MultiModalPredictor_90598040142139
// MI455X (gfx1250) — matrix operand order repaired and checked by reading, not compiled, not run on hardware
//
#include <hip/hip_runtime.h>
#include <hip/hip_bf16.h>

typedef __bf16 v16bf __attribute__((ext_vector_type(16)));
typedef float  v8f   __attribute__((ext_vector_type(8)));

#define HD 128
#define INP 276
#define KPAD 288

union AFrag { unsigned int u[8]; v16bf v; };

#define WMMA_BF16(A_, B_, C_) \
  __builtin_amdgcn_wmma_f32_16x16x32_bf16(false, (A_), false, (B_), (short)0, (C_), false, false)

static __device__ __forceinline__ float sigmoidf_(float v) {
  return 1.0f / (1.0f + __expf(-v));
}

// ---------------- prep kernels ----------------

// W1: [276][128] f32 row-major -> packed bf16 [9][8][32][16], K padded to 288
__global__ void pack_w1_k(const float* __restrict__ W, __hip_bfloat16* __restrict__ out) {
  int i = blockIdx.x * 256 + threadIdx.x;
  if (i >= 9 * 8 * 32 * 16) return;
  int e = i & 15, lane = (i >> 4) & 31, nt = (i >> 9) & 7, kc = i >> 12;
  int n = nt * 16 + (lane & 15);
  int k = kc * 32 + ((e < 8) ? ((lane >> 4) * 8 + e) : (16 + (lane >> 4) * 8 + (e - 8)));
  float v = (k < INP) ? W[k * HD + n] : 0.0f;
  out[i] = __float2bfloat16(v);
}

// W2: [128][128] f32 row-major -> packed bf16 [4][8][32][16]
__global__ void pack_w2_k(const float* __restrict__ W, __hip_bfloat16* __restrict__ out) {
  int i = blockIdx.x * 256 + threadIdx.x;
  if (i >= 4 * 8 * 32 * 16) return;
  int e = i & 15, lane = (i >> 4) & 31, nt = (i >> 9) & 7, kc = i >> 12;
  int n = nt * 16 + (lane & 15);
  int k = kc * 32 + ((e < 8) ? ((lane >> 4) * 8 + e) : (16 + (lane >> 4) * 8 + (e - 8)));
  out[i] = __float2bfloat16(W[k * HD + n]);
}

// GRU weight [384][128] row-major; B = W^T (128x384) -> packed bf16 [4][24][32][16]
__global__ void pack_gruT_k(const float* __restrict__ W, __hip_bfloat16* __restrict__ out) {
  int i = blockIdx.x * 256 + threadIdx.x;
  if (i >= 4 * 24 * 32 * 16) return;
  int e = i & 15, lane = (i >> 4) & 31;
  int t = i >> 9;
  int nt = t % 24, kc = t / 24;
  int n = nt * 16 + (lane & 15);               // 0..383
  int k = kc * 32 + ((e < 8) ? ((lane >> 4) * 8 + e) : (16 + (lane >> 4) * 8 + (e - 8))); // 0..127
  out[i] = __float2bfloat16(W[n * HD + k]);     // transpose
}

__global__ void conv_x_k(const float* __restrict__ x, __hip_bfloat16* __restrict__ o, long n) {
  long i = (long)blockIdx.x * 256 + threadIdx.x;
  long stride = (long)gridDim.x * 256;
  for (; i < n; i += stride) o[i] = __float2bfloat16(x[i]);
}

__global__ void zero_f_k(float* __restrict__ p, long n) {
  long i = (long)blockIdx.x * 256 + threadIdx.x;
  long stride = (long)gridDim.x * 256;
  for (; i < n; i += stride) p[i] = 0.0f;
}

// ---------------- edge kernel: message MLP + scatter ----------------
// 128 threads = 4 waves, 64 edges per block.
__global__ __launch_bounds__(128)
void edge_kernel(const __hip_bfloat16* __restrict__ xbf,   // [N][128] bf16
                 const float* __restrict__ eattr,          // [E][16]
                 const float* __restrict__ pos,            // [N][3]
                 const float* __restrict__ b1,
                 const float* __restrict__ g1,
                 const float* __restrict__ be1,
                 const float* __restrict__ b2,
                 const __hip_bfloat16* __restrict__ W1p,   // [9][8][32][16]
                 const __hip_bfloat16* __restrict__ W2p,   // [4][8][32][16]
                 const int* __restrict__ srcI,
                 const int* __restrict__ dstI,
                 float* __restrict__ sums,                 // [N][128]
                 float* __restrict__ cnt,                  // [N]
                 int E_) {
  __shared__ __align__(16) unsigned char smem[49152];
  __hip_bfloat16* A  = (__hip_bfloat16*)smem;            // [64][288] bf16 (GEMM1 A stage)
  float* Cs          = (float*)smem;                     // [64][128] f32 (reuses A region)
  __hip_bfloat16* O  = (__hip_bfloat16*)(smem + 32768);  // [64][128] bf16 (LN+SiLU out)

  const int tid = threadIdx.x;
  const int e0 = blockIdx.x * 64;

  // geometry + edge_attr + pad columns (one thread per edge row)
  if (tid < 64) {
    int e = e0 + tid;
    int ec = (e < E_) ? e : (E_ - 1);
    int s = srcI[ec], d = dstI[ec];
    float rx = (pos[s * 3 + 0] - pos[d * 3 + 0]) * 0.2f;
    float ry = (pos[s * 3 + 1] - pos[d * 3 + 1]) * 0.2f;
    float rz = (pos[s * 3 + 2] - pos[d * 3 + 2]) * 0.2f;
    float d2 = rx * rx + ry * ry + rz * rz;
    __hip_bfloat16* row = A + (size_t)tid * KPAD;
#pragma unroll
    for (int j = 0; j < 16; ++j) row[256 + j] = __float2bfloat16(eattr[(size_t)ec * 16 + j]);
    row[272] = __float2bfloat16(rx);
    row[273] = __float2bfloat16(ry);
    row[274] = __float2bfloat16(rz);
    row[275] = __float2bfloat16(d2);
#pragma unroll
    for (int j = INP; j < KPAD; ++j) row[j] = __float2bfloat16(0.0f);
    if (e < E_) atomicAdd(&cnt[d], 1.0f);
  }
  // gather x[dst] (cols 0..127) and x[src] (cols 128..255), 32 threads per row
  {
    int rg = tid >> 5, l32 = tid & 31;
    for (int r = rg; r < 64; r += 4) {
      int e = e0 + r;
      int ec = (e < E_) ? e : (E_ - 1);
      int d = dstI[ec], s = srcI[ec];
      const unsigned int* xd = (const unsigned int*)(xbf + (size_t)d * HD);
      const unsigned int* xs = (const unsigned int*)(xbf + (size_t)s * HD);
      unsigned int* rp = (unsigned int*)(A + (size_t)r * KPAD);
      rp[l32]      = xd[l32];
      rp[l32 + 32] = xd[l32 + 32];
      rp[l32 + 64] = xs[l32];
      rp[l32 + 96] = xs[l32 + 32];
    }
  }
  __syncthreads();

  const int l = tid & 31;
  const int wv = tid >> 5;
  const int rowbase = wv * 16;
  const int m  = rowbase + (l & 15);            // A-matrix row for this lane
  const int kbU = (l < 16) ? 0 : 4;             // dword offset per A-layout
  const int n0 = l & 15;                        // C col within tile
  const int mr = rowbase + ((l < 16) ? 0 : 8);  // C row base for this lane

  // GEMM1: [64x288] x [288x128]
  v8f acc[8];
#pragma unroll
  for (int t = 0; t < 8; ++t) acc[t] = (v8f){0, 0, 0, 0, 0, 0, 0, 0};
  const unsigned int* Ar = (const unsigned int*)(A + (size_t)m * KPAD);
  for (int kc = 0; kc < 9; ++kc) {
    AFrag a;
    int k0 = kc * 16;
#pragma unroll
    for (int j = 0; j < 4; ++j) {
      a.u[j]     = Ar[k0 + kbU + j];
      a.u[4 + j] = Ar[k0 + 8 + kbU + j];
    }
#pragma unroll
    for (int nt = 0; nt < 8; ++nt) {
      const v16bf* bp = (const v16bf*)(W1p + (((size_t)(kc * 8 + nt) * 32 + l) * 16));
      acc[nt] = WMMA_BF16(a.v, *bp, acc[nt]);
    }
  }
  __syncthreads();  // all waves done reading A before overwriting with Cs

  // spill C to LDS (C layout: lane 0-15 col, VGPR r row; lanes 16-31 rows +8)
#pragma unroll
  for (int nt = 0; nt < 8; ++nt)
#pragma unroll
    for (int r = 0; r < 8; ++r)
      Cs[(size_t)(mr + r) * HD + nt * 16 + n0] = acc[nt][r];
  __syncthreads();

  // rowwise LayerNorm + SiLU -> bf16
  if (tid < 64) {
    const float* cr = Cs + (size_t)tid * HD;
    float s1 = 0.0f, s2 = 0.0f;
    for (int j = 0; j < HD; ++j) { float v = cr[j] + b1[j]; s1 += v; s2 += v * v; }
    float mu = s1 * (1.0f / HD);
    float var = s2 * (1.0f / HD) - mu * mu;
    float inv = rsqrtf(var + 1e-5f);
    __hip_bfloat16* orow = O + (size_t)tid * HD;
    for (int j = 0; j < HD; ++j) {
      float t = g1[j] * ((cr[j] + b1[j]) - mu) * inv + be1[j];
      orow[j] = __float2bfloat16(t * sigmoidf_(t));
    }
  }
  __syncthreads();

  // GEMM2: [64x128] x [128x128]
  v8f acc2[8];
#pragma unroll
  for (int t = 0; t < 8; ++t) acc2[t] = (v8f){0, 0, 0, 0, 0, 0, 0, 0};
  const unsigned int* Ar2 = (const unsigned int*)(O + (size_t)m * HD);
  for (int kc = 0; kc < 4; ++kc) {
    AFrag a;
    int k0 = kc * 16;
#pragma unroll
    for (int j = 0; j < 4; ++j) {
      a.u[j]     = Ar2[k0 + kbU + j];
      a.u[4 + j] = Ar2[k0 + 8 + kbU + j];
    }
#pragma unroll
    for (int nt = 0; nt < 8; ++nt) {
      const v16bf* bp = (const v16bf*)(W2p + (((size_t)(kc * 8 + nt) * 32 + l) * 16));
      acc2[nt] = WMMA_BF16(a.v, *bp, acc2[nt]);
    }
  }

  // scatter-add (mean numerator) with bias
  int dd[8], ok[8];
#pragma unroll
  for (int r = 0; r < 8; ++r) {
    int e = e0 + mr + r;
    ok[r] = (e < E_);
    dd[r] = dstI[ok[r] ? e : (E_ - 1)];
  }
#pragma unroll
  for (int nt = 0; nt < 8; ++nt) {
    int col = nt * 16 + n0;
    float bc = b2[col];
#pragma unroll
    for (int r = 0; r < 8; ++r) {
      if (ok[r]) atomicAdd(&sums[(size_t)dd[r] * HD + col], acc2[nt][r] + bc);
    }
  }
}

// ---------------- node kernel: mean + GRU + LN ----------------
// 64 threads = 2 waves, 32 node rows per block.
__global__ __launch_bounds__(64)
void node_kernel(const float* __restrict__ sums,
                 const float* __restrict__ cnt,
                 const float* __restrict__ x,
                 const float* __restrict__ bih,
                 const float* __restrict__ bhh,
                 const float* __restrict__ g2,
                 const float* __restrict__ be2,
                 const __hip_bfloat16* __restrict__ WihTp,  // [4][24][32][16]
                 const __hip_bfloat16* __restrict__ WhhTp,  // [4][24][32][16]
                 float* __restrict__ out,
                 int N_) {
  __shared__ __align__(16) unsigned char smem[32768];
  __hip_bfloat16* aggS = (__hip_bfloat16*)smem;            // [32][128]
  __hip_bfloat16* xS   = (__hip_bfloat16*)(smem + 8192);   // [32][128]
  float* preS          = (float*)(smem + 16384);           // [32][128]

  const int tid = threadIdx.x;
  const int r0 = blockIdx.x * 32;

  for (int i = tid; i < 32 * HD; i += 64) {
    int r = i >> 7, c = i & 127;
    int row = r0 + r; if (row >= N_) row = N_ - 1;
    float cn = cnt[row]; if (cn < 1.0f) cn = 1.0f;
    aggS[i] = __float2bfloat16(sums[(size_t)row * HD + c] / cn);
    xS[i]   = __float2bfloat16(x[(size_t)row * HD + c]);
  }
  __syncthreads();

  const int l = tid & 31;
  const int wv = tid >> 5;
  const int rowbase = wv * 16;
  const int m  = rowbase + (l & 15);
  const int kbU = (l < 16) ? 0 : 4;
  const int n0 = l & 15;
  const int mr = rowbase + ((l < 16) ? 0 : 8);

  const unsigned int* Ag = (const unsigned int*)(aggS + (size_t)m * HD);
  const unsigned int* Ax = (const unsigned int*)(xS + (size_t)m * HD);

  for (int nt = 0; nt < 8; ++nt) {
    v8f gr = (v8f){0,0,0,0,0,0,0,0}, gz = gr, gn = gr, hr = gr, hz = gr, hn = gr;
    for (int kc = 0; kc < 4; ++kc) {
      AFrag a, b;
      int k0 = kc * 16;
#pragma unroll
      for (int j = 0; j < 4; ++j) {
        a.u[j]     = Ag[k0 + kbU + j];
        a.u[4 + j] = Ag[k0 + 8 + kbU + j];
        b.u[j]     = Ax[k0 + kbU + j];
        b.u[4 + j] = Ax[k0 + 8 + kbU + j];
      }
      const v16bf* br = (const v16bf*)(WihTp + (((size_t)(kc * 24 + nt)      * 32 + l) * 16));
      const v16bf* bz = (const v16bf*)(WihTp + (((size_t)(kc * 24 + nt + 8)  * 32 + l) * 16));
      const v16bf* bn = (const v16bf*)(WihTp + (((size_t)(kc * 24 + nt + 16) * 32 + l) * 16));
      const v16bf* cr = (const v16bf*)(WhhTp + (((size_t)(kc * 24 + nt)      * 32 + l) * 16));
      const v16bf* cz = (const v16bf*)(WhhTp + (((size_t)(kc * 24 + nt + 8)  * 32 + l) * 16));
      const v16bf* cn2 = (const v16bf*)(WhhTp + (((size_t)(kc * 24 + nt + 16) * 32 + l) * 16));
      gr = WMMA_BF16(a.v, *br, gr);
      gz = WMMA_BF16(a.v, *bz, gz);
      gn = WMMA_BF16(a.v, *bn, gn);
      hr = WMMA_BF16(b.v, *cr, hr);
      hz = WMMA_BF16(b.v, *cz, hz);
      hn = WMMA_BF16(b.v, *cn2, hn);
    }
    int col = nt * 16 + n0;
    float bir = bih[col], biz = bih[HD + col], bin = bih[2 * HD + col];
    float bhr = bhh[col], bhz = bhh[HD + col], bhn = bhh[2 * HD + col];
#pragma unroll
    for (int r = 0; r < 8; ++r) {
      int row = r0 + mr + r;
      int rc = (row < N_) ? row : (N_ - 1);
      float rrg = sigmoidf_((gr[r] + bir) + (hr[r] + bhr));
      float zzg = sigmoidf_((gz[r] + biz) + (hz[r] + bhz));
      float nng = tanhf((gn[r] + bin) + rrg * (hn[r] + bhn));
      float xv = x[(size_t)rc * HD + col];
      float upd = (1.0f - zzg) * nng + zzg * xv;
      preS[(size_t)(mr + r) * HD + col] = xv + upd;
    }
  }
  __syncthreads();

  if (tid < 32) {
    int row = r0 + tid;
    if (row < N_) {
      const float* p = preS + (size_t)tid * HD;
      float s1 = 0.0f, s2 = 0.0f;
      for (int j = 0; j < HD; ++j) { float v = p[j]; s1 += v; s2 += v * v; }
      float mu = s1 * (1.0f / HD);
      float var = s2 * (1.0f / HD) - mu * mu;
      float inv = rsqrtf(var + 1e-5f);
      float* o = out + (size_t)row * HD;
      for (int j = 0; j < HD; ++j) o[j] = g2[j] * (p[j] - mu) * inv + be2[j];
    }
  }
}

// ---------------- launch ----------------

extern "C" void kernel_launch(void* const* d_in, const int* in_sizes, int n_in,
                              void* d_out, int out_size, void* d_ws, size_t ws_size,
                              hipStream_t stream) {
  const float* x     = (const float*)d_in[0];
  const float* eattr = (const float*)d_in[1];
  const float* pos   = (const float*)d_in[2];
  const float* W1    = (const float*)d_in[3];
  const float* b1    = (const float*)d_in[4];
  const float* ln1g  = (const float*)d_in[5];
  const float* ln1b  = (const float*)d_in[6];
  const float* W2    = (const float*)d_in[7];
  const float* b2    = (const float*)d_in[8];
  const float* Wih   = (const float*)d_in[9];
  const float* bih   = (const float*)d_in[10];
  const float* Whh   = (const float*)d_in[11];
  const float* bhh   = (const float*)d_in[12];
  const float* ln2g  = (const float*)d_in[13];
  const float* ln2b  = (const float*)d_in[14];
  const int*   eidx  = (const int*)d_in[15];
  float* out = (float*)d_out;

  const int N_ = in_sizes[0] / HD;
  const int E_ = in_sizes[15] / 2;
  const int* srcI = eidx;        // edge_index[0]
  const int* dstI = eidx + E_;   // edge_index[1]

  char* ws = (char*)d_ws;
  __hip_bfloat16* W1p   = (__hip_bfloat16*)(ws + 0);                        // 73,728 B
  __hip_bfloat16* W2p   = (__hip_bfloat16*)(ws + 73728);                    // 32,768 B
  __hip_bfloat16* WihTp = (__hip_bfloat16*)(ws + 106496);                   // 98,304 B
  __hip_bfloat16* WhhTp = (__hip_bfloat16*)(ws + 204800);                   // 98,304 B
  __hip_bfloat16* xbf   = (__hip_bfloat16*)(ws + 303104);                   // N*128*2
  size_t sums_off = 303104 + (size_t)N_ * HD * 2;
  float* sums = (float*)(ws + sums_off);                                    // N*128*4
  float* cnt  = (float*)(ws + sums_off + (size_t)N_ * HD * 4);              // N*4

  // prep: weight packing, x conversion, zeroing scratch (every call — ws not re-poisoned)
  pack_w1_k<<<(9 * 8 * 32 * 16 + 255) / 256, 256, 0, stream>>>(W1, W1p);
  pack_w2_k<<<(4 * 8 * 32 * 16 + 255) / 256, 256, 0, stream>>>(W2, W2p);
  pack_gruT_k<<<(4 * 24 * 32 * 16 + 255) / 256, 256, 0, stream>>>(Wih, WihTp);
  pack_gruT_k<<<(4 * 24 * 32 * 16 + 255) / 256, 256, 0, stream>>>(Whh, WhhTp);
  long nx = (long)N_ * HD;
  conv_x_k<<<(int)((nx + 255) / 256), 256, 0, stream>>>(x, xbf, nx);
  long nz = (long)N_ * (HD + 1);  // sums + cnt contiguous
  zero_f_k<<<(int)((nz + 255) / 256), 256, 0, stream>>>(sums, nz);

  edge_kernel<<<(E_ + 63) / 64, 128, 0, stream>>>(
      xbf, eattr, pos, b1, ln1g, ln1b, b2, W1p, W2p, srcI, dstI, sums, cnt, E_);

  node_kernel<<<(N_ + 31) / 32, 64, 0, stream>>>(
      sums, cnt, x, bih, bhh, ln2g, ln2b, WihTp, WhhTp, out, N_);
}
